// CIN_28166395527380
// MI455X (gfx1250) — matrix operand order repaired and checked by reading, not compiled, not run on hardware
//
#include <hip/hip_runtime.h>

// ---------------------------------------------------------------------------
// CIN (xDeepFM cross-interaction) for MI455X / gfx1250.
//
// z[b,o,d] = sum_{f,g} W[o,f,g] * x0[b,f,d] * h[b,g,d]   (+bias, relu)
// Factorized per (b, f):  S_f[d,o] = sum_g h[b,g,d] * W[o,f,g]   <- bf16 WMMA
//                         z[d,o]  += x0[b,f,d] * S_f[d,o]        <- VALU fma
// ~2.2e11 FLOPs vs ~25 MB of data => compute bound => bf16 WMMA, f32 accum.
//
// v2: 4 batch elements per workgroup (B-tile register reuse => W L2 traffic
// /4), A staged in LDS via GLOBAL_LOAD_ASYNC_TO_LDS_B128 (ASYNCcnt), B tiles
// double-buffered in registers to keep v_wmma off the load-latency path.
// ---------------------------------------------------------------------------

typedef __bf16  v16bf __attribute__((ext_vector_type(16)));
typedef float   v8f   __attribute__((ext_vector_type(8)));

#define B_SZ   2048
#define F0_SZ  39
#define D_SZ   16
#define NPAD   256     // padded output-channel dim (13 real 16-tiles -> 16)
#define COUT   200
#define MB     4       // batch elements per workgroup

__device__ __forceinline__ v8f v8zero() {
  v8f z = {0.f, 0.f, 0.f, 0.f, 0.f, 0.f, 0.f, 0.f};
  return z;
}

// Build a v16bf from two 16-byte-aligned 8-element chunks (global).
__device__ __forceinline__ v16bf ld16x2(const __bf16* lo, const __bf16* hi) {
  union { v16bf v; uint4 q[2]; } u;
  u.q[0] = *(const uint4*)lo;   // elements 0..7
  u.q[1] = *(const uint4*)hi;   // elements 8..15
  return u.v;
}

// Same but from LDS (byte offsets into the shared block).
__device__ __forceinline__ v16bf ld16x2_lds(const char* p) {
  union { v16bf v; uint4 q[2]; } u;
  u.q[0] = *(const uint4*)(p);        // elements 0..7   (K+0..7)
  u.q[1] = *(const uint4*)(p + 16);   // elements 8..15  (K+8..15)
  return u.v;
}

// ---- prep: x (B,F0,D) f32 -> Xt[b][d][64] bf16 (f zero-padded to 64) -------
__global__ void cvt_x_kernel(const float* __restrict__ x, __bf16* __restrict__ xt) {
  int idx = blockIdx.x * blockDim.x + threadIdx.x;          // < B*16*64
  int k = idx & 63;
  int m = idx >> 6;            // (b,d) row
  int b = m >> 4;
  int d = m & 15;
  float v = (k < F0_SZ) ? x[((size_t)b * F0_SZ + k) * D_SZ + d] : 0.f;
  xt[idx] = (__bf16)v;
}

// ---- prep: W (COUT,F0,G) f32 -> Wc[256][F0][KP] bf16 (zero padded) ---------
__global__ void cvt_w_kernel(const float* __restrict__ W, __bf16* __restrict__ Wc,
                             int G, int KP, int total) {
  int idx = blockIdx.x * blockDim.x + threadIdx.x;
  if (idx >= total) return;
  int kp   = idx % KP;
  int rest = idx / KP;
  int f = rest % F0_SZ;
  int o = rest / F0_SZ;
  float v = (o < COUT && kp < G) ? W[((size_t)o * F0_SZ + f) * G + kp] : 0.f;
  Wc[idx] = (__bf16)v;
}

// ---- main layer kernel -----------------------------------------------------
// Workgroup: 256 threads (8 waves), MB=4 batch elements (4 M-tiles of d=0..15).
// Wave w owns output columns [32w, 32w+32) (2 N-tiles), reusing each B tile
// register pair across all 4 M-tiles. NC = K chunks of 32 (NC=2 or 7).
template <int NC>
__global__ void __launch_bounds__(256)
cin_layer_kernel(const __bf16* __restrict__ Hin,   // [B*16][NC*32] bf16
                 const __bf16* __restrict__ Wc,    // [256][F0][NC*32] bf16
                 const float*  __restrict__ x,     // original x, f32 (scaling)
                 const float*  __restrict__ bias,  // [COUT]
                 __bf16* __restrict__ Hnext,       // [B*16][KPnext] or null
                 int KPnext,
                 float* __restrict__ out, int outOff) {
  constexpr int KP   = NC * 32;        // K (g) padded
  constexpr int KPL  = KP + 8;         // LDS row stride (elems): bank-spread
  constexpr int ABYTES = MB * 16 * KPL * 2;            // A staging
  constexpr int XOFF   = ABYTES;                        // x0 f32 region
  constexpr int CH   = KP / 8;         // 16B chunks per A row

  __shared__ __align__(16) char smem[ABYTES + MB * 16 * 40 * 4];

  const int tid  = threadIdx.x;
  const int lane = tid & 31;
  const int w    = tid >> 5;
  const int ln   = lane & 15;     // A row (=d) / B col / C col
  const int hi   = lane >> 4;     // half-wave selector
  const int bb0  = blockIdx.x * MB;

  // ---- stage A (Ht rows for MB batch elems) into LDS, async path ----------
  {
    const __bf16* src = Hin + (size_t)bb0 * 16 * KP;
    for (int i = tid; i < MB * 16 * CH; i += 256) {
      int row = i / CH;          // (mb*16 + d)
      int cc  = i % CH;          // 16B chunk within row
      unsigned ldsOff = (unsigned)(size_t)(smem + row * (KPL * 2) + cc * 16);
      unsigned long long gaddr =
          (unsigned long long)(const void*)(src + (size_t)row * KP + cc * 8);
      asm volatile("global_load_async_to_lds_b128 %0, %1, off"
                   :: "v"(ldsOff), "v"(gaddr) : "memory");
    }
  }
  // ---- stage x0 (f32, [mb][d][40]) for the per-row scaling ----------------
  {
    float* xp = (float*)(smem + XOFF);
    for (int i = tid; i < MB * F0_SZ * D_SZ; i += 256) {
      int mb = i / (F0_SZ * D_SZ);
      int r  = i % (F0_SZ * D_SZ);
      int f = r / D_SZ, d = r % D_SZ;
      xp[(mb * 16 + d) * 40 + f] =
          x[((size_t)(bb0 + mb) * F0_SZ + f) * D_SZ + d];
    }
  }
  asm volatile("s_wait_asynccnt 0" ::: "memory");
  __syncthreads();

  // ---- per-wave constants --------------------------------------------------
  const int o0 = w * 32 + ln;           // column of N-tile 0
  const int o1 = o0 + 16;               // column of N-tile 1
  const __bf16* wp0 = Wc + (size_t)o0 * F0_SZ * KP + hi * 16;
  const __bf16* wp1 = Wc + (size_t)o1 * F0_SZ * KP + hi * 16;
  float biasv[2];
  biasv[0] = (o0 < COUT) ? bias[o0] : 0.f;
  biasv[1] = (o1 < COUT) ? bias[o1] : 0.f;
  const float* xp = (const float*)(smem + XOFF);
  // per-lane base byte offset into A LDS rows (row = mb*16 + ln, +hi*16 elems)
  const char* aBase = smem + ((size_t)ln * KPL + hi * 16) * 2;

  v8f Z[MB][2];
#pragma unroll
  for (int mb = 0; mb < MB; ++mb) { Z[mb][0] = v8zero(); Z[mb][1] = v8zero(); }

  for (int f = 0; f < F0_SZ; ++f) {
    v8f S[MB][2];
#pragma unroll
    for (int mb = 0; mb < MB; ++mb) { S[mb][0] = v8zero(); S[mb][1] = v8zero(); }

    // two-stage double buffer for B tiles: loads for chunk c+1 fly over the
    // 8 WMMAs of chunk c.
    v16bf Bv[2][2];
    {
      const __bf16* p0 = wp0 + (size_t)f * KP;
      const __bf16* p1 = wp1 + (size_t)f * KP;
      Bv[0][0] = ld16x2(p0, p0 + 8);
      Bv[0][1] = ld16x2(p1, p1 + 8);
    }
#pragma unroll
    for (int c = 0; c < NC; ++c) {
      const int cur = c & 1;
      if (c + 1 < NC) {
        const __bf16* p0 = wp0 + (size_t)f * KP + (c + 1) * 32;
        const __bf16* p1 = wp1 + (size_t)f * KP + (c + 1) * 32;
        Bv[cur ^ 1][0] = ld16x2(p0, p0 + 8);
        Bv[cur ^ 1][1] = ld16x2(p1, p1 + 8);
      }
#pragma unroll
      for (int mb = 0; mb < MB; ++mb) {
        v16bf Av = ld16x2_lds(aBase + mb * (16 * KPL * 2) + c * 64);
        S[mb][0] = __builtin_amdgcn_wmma_f32_16x16x32_bf16(
            false, Av, false, Bv[cur][0], (short)0, S[mb][0], false, false);
        S[mb][1] = __builtin_amdgcn_wmma_f32_16x16x32_bf16(
            false, Av, false, Bv[cur][1], (short)0, S[mb][1], false, false);
      }
    }

    // z[d,o] += x0[b,f,d] * S_f[d,o]   (row d = r + 8*hi; broadcast LDS read)
#pragma unroll
    for (int mb = 0; mb < MB; ++mb) {
#pragma unroll
      for (int r = 0; r < 8; ++r) {
        float s = xp[(mb * 16 + hi * 8 + r) * 40 + f];
        Z[mb][0][r] += s * S[mb][0][r];
        Z[mb][1][r] += s * S[mb][1][r];
      }
    }
  }

  // ---- epilogue: bias + relu, next-layer activations, d-pooled output -----
#pragma unroll
  for (int mb = 0; mb < MB; ++mb) {
    const int bcur = bb0 + mb;
#pragma unroll
    for (int tt = 0; tt < 2; ++tt) {
      const int o = w * 32 + tt * 16 + ln;
      float csum = 0.f;
#pragma unroll
      for (int r = 0; r < 8; ++r) {
        float h = Z[mb][tt][r] + biasv[tt];
        h = h > 0.f ? h : 0.f;
        if (o < COUT) csum += h;
        if (Hnext != nullptr && o < KPnext) {
          int d = hi * 8 + r;
          __bf16 hv = (o < COUT) ? (__bf16)h : (__bf16)0.f;  // zero K-pad
          Hnext[((size_t)bcur * 16 + d) * KPnext + o] = hv;
        }
      }
      // combine the two half-wave row groups (rows 0..7 with rows 8..15)
      csum += __shfl_xor(csum, 16, 32);
      if (hi == 0 && o < COUT)
        out[(size_t)bcur * 600 + outOff + o] = csum;
    }
  }
}

// ---------------------------------------------------------------------------
extern "C" void kernel_launch(void* const* d_in, const int* in_sizes, int n_in,
                              void* d_out, int out_size, void* d_ws, size_t ws_size,
                              hipStream_t stream) {
  const float* x  = (const float*)d_in[0];
  const float* W0 = (const float*)d_in[1];
  const float* b0 = (const float*)d_in[2];
  const float* W1 = (const float*)d_in[3];
  const float* b1 = (const float*)d_in[4];
  const float* W2 = (const float*)d_in[5];
  const float* b2 = (const float*)d_in[6];
  float* out = (float*)d_out;
  char* ws = (char*)d_ws;

  // workspace layout (all offsets 256B-aligned)
  const size_t WC0_BYTES  = (size_t)NPAD * F0_SZ * 64 * 2;    // 1,277,952
  const size_t WC12_BYTES = (size_t)NPAD * F0_SZ * 224 * 2;   // 4,472,832
  const size_t XT_BYTES   = (size_t)B_SZ * 16 * 64 * 2;       // 4,194,304
  const size_t H_BYTES    = (size_t)B_SZ * 16 * 224 * 2;      // 14,680,064

  __bf16* Wc0 = (__bf16*)(ws);
  __bf16* Wc1 = (__bf16*)(ws + WC0_BYTES);
  __bf16* Wc2 = (__bf16*)(ws + WC0_BYTES + WC12_BYTES);
  __bf16* Xt  = (__bf16*)(ws + WC0_BYTES + 2 * WC12_BYTES);
  __bf16* H1  = (__bf16*)(ws + WC0_BYTES + 2 * WC12_BYTES + XT_BYTES);
  __bf16* H2  = (__bf16*)(ws + WC0_BYTES + 2 * WC12_BYTES + XT_BYTES + H_BYTES);

  // precision conversion / layout packing
  {
    int total = NPAD * F0_SZ * 64;      // layer 0 weights (G=39 -> KP=64)
    cvt_w_kernel<<<(total + 255) / 256, 256, 0, stream>>>(W0, Wc0, 39, 64, total);
  }
  {
    int total = NPAD * F0_SZ * 224;     // layers 1/2 (G=200 -> KP=224)
    cvt_w_kernel<<<(total + 255) / 256, 256, 0, stream>>>(W1, Wc1, 200, 224, total);
    cvt_w_kernel<<<(total + 255) / 256, 256, 0, stream>>>(W2, Wc2, 200, 224, total);
  }
  {
    int total = B_SZ * 16 * 64;
    cvt_x_kernel<<<total / 256, 256, 0, stream>>>(x, Xt);
  }

  // three CIN layers; each writes its 200-col slice of out and the next H
  cin_layer_kernel<2><<<B_SZ / MB, 256, 0, stream>>>(Xt, Wc0, x, b0, H1, 224, out, 0);
  cin_layer_kernel<7><<<B_SZ / MB, 256, 0, stream>>>(H1, Wc1, x, b1, H2, 224, out, 200);
  cin_layer_kernel<7><<<B_SZ / MB, 256, 0, stream>>>(H2, Wc2, x, b2, nullptr, 224, out, 400);
}
